// GraphSAGE_local_94489281444
// MI455X (gfx1250) — matrix operand order repaired and checked by reading, not compiled, not run on hardware
//
#include <hip/hip_runtime.h>
#include <hip/hip_bf16.h>
#include <stdint.h>

// ---------------------------------------------------------------------------
// GraphSAGE (2-layer, project=True) for MI455X / gfx1250 (wave32, WMMA).
// GEMMs in bf16 via v_wmma_f32_16x16x32_bf16 (fp32 accumulate).
// Edge aggregation via bf16 gather + hardware global_atomic_add_f32 scatter.
// ---------------------------------------------------------------------------

typedef __attribute__((ext_vector_type(16))) __bf16 v16bf;
typedef __attribute__((ext_vector_type(8)))  float  v8f;

union Frag16 {
  v16bf v;
  uint4 q[2];
};

__device__ __forceinline__ unsigned int f32_to_bf16_rne(float f) {
  union { float f; unsigned int u; } c; c.f = f;
  unsigned int u = c.u;
  return (u + 0x7FFFu + ((u >> 16) & 1u)) >> 16;
}
__device__ __forceinline__ unsigned int pack_bf16x2(float lo, float hi) {
  return f32_to_bf16_rne(lo) | (f32_to_bf16_rne(hi) << 16);
}
__device__ __forceinline__ float bf16_to_f32(unsigned int bits16) {
  union { unsigned int u; float f; } c; c.u = bits16 << 16;
  return c.f;
}

// Hardware fp32 atomic add (no-return form), device scope so L2 atomic units
// merge updates across WGPs. Avoids any CAS-loop lowering of atomicAdd(float*).
__device__ __forceinline__ void gatomic_add_f32(float* p, float v) {
  asm volatile("global_atomic_add_f32 %0, %1, off scope:SCOPE_DEV"
               :
               : "v"(p), "v"(v)
               : "memory");
}

// ---------------------------------------------------------------------------
// fp32 -> bf16 conversion, 4 elems/thread (float4 load, packed uint2 store).
// All tensor sizes here are multiples of 128, so n % 4 == 0.
// ---------------------------------------------------------------------------
__global__ void __launch_bounds__(256)
k_f32_to_bf16(const float* __restrict__ in, unsigned short* __restrict__ out, int nquads) {
  int i = blockIdx.x * 256 + threadIdx.x;
  if (i >= nquads) return;
  const float4 v = *(const float4*)(in + (size_t)i * 4);
  uint2 pk;
  pk.x = pack_bf16x2(v.x, v.y);
  pk.y = pack_bf16x2(v.z, v.w);
  *(uint2*)(out + (size_t)i * 4) = pk;
}

// ---------------------------------------------------------------------------
// Projection GEMM: out_bf16[M,128] = bf16(relu(A[M,128] @ W[128,128]^T + b))
// One wave per 16x16 tile; 8 waves/block cover all 128 output features.
// K = 128 -> 4 WMMA per tile. Fragments loaded directly (both operands are
// K-contiguous; weights stay L2-resident).
// ---------------------------------------------------------------------------
__global__ void __launch_bounds__(256)
k_gemm_proj(const unsigned short* __restrict__ A, const unsigned short* __restrict__ W,
            const float* __restrict__ bias, unsigned short* __restrict__ outb) {
  const int lane  = threadIdx.x & 31;
  const int ntile = threadIdx.x >> 5;          // 0..7
  const int mtile = blockIdx.x;
  const int m     = lane & 15;
  const int half  = lane >> 4;

  const unsigned short* arow = A + (size_t)(mtile * 16 + m) * 128;
  const int n = ntile * 16 + (lane & 15);
  const unsigned short* wrow = W + (size_t)n * 128;

  v8f acc = {};
#pragma unroll
  for (int k0 = 0; k0 < 128; k0 += 32) {
    __builtin_prefetch(arow + k0 + 32, 0, 3);
    Frag16 a, b;
    // 16-bit A 16x32 lane layout: lane holds K = half*8 + {0..7, 16..23}
    a.q[0] = *(const uint4*)(arow + k0 + half * 8);
    a.q[1] = *(const uint4*)(arow + k0 + half * 8 + 16);
    // 16-bit B 32x16 lane layout: lane holds col (lane&15), K = half*8 + {0..7, 16..23}
    b.q[0] = *(const uint4*)(wrow + k0 + half * 8);
    b.q[1] = *(const uint4*)(wrow + k0 + half * 8 + 16);
    acc = __builtin_amdgcn_wmma_f32_16x16x32_bf16(false, a.v, false, b.v,
                                                  (short)0, acc, false, false);
  }

  const float bv = bias[n];
#pragma unroll
  for (int r = 0; r < 8; ++r) {
    const int row = mtile * 16 + half * 8 + r;   // C/D layout: M = r + half*8, N = lane&15
    float v = acc[r] + bv;
    v = v > 0.f ? v : 0.f;                       // project=True applies relu
    outb[(size_t)row * 128 + n] = (unsigned short)f32_to_bf16_rne(v);
  }
}

// ---------------------------------------------------------------------------
// Dual GEMM: out = A1@W1^T + A2@W2^T + b  (K = 128, NOUT = NTILES*16)
// Layer 1: relu + bf16 out.  Layer 2: raw fp32 out (normalize done later).
// ---------------------------------------------------------------------------
template <int NTILES>
__global__ void __launch_bounds__(NTILES * 32)
k_dual_gemm(const unsigned short* __restrict__ A1, const unsigned short* __restrict__ W1,
            const unsigned short* __restrict__ A2, const unsigned short* __restrict__ W2,
            const float* __restrict__ bias,
            unsigned short* __restrict__ outb, float* __restrict__ outf, int do_relu) {
  const int lane  = threadIdx.x & 31;
  const int ntile = threadIdx.x >> 5;
  const int mtile = blockIdx.x;
  const int m     = lane & 15;
  const int half  = lane >> 4;
  const int NOUT  = NTILES * 16;

  const unsigned short* a1row = A1 + (size_t)(mtile * 16 + m) * 128;
  const unsigned short* a2row = A2 + (size_t)(mtile * 16 + m) * 128;
  const int n = ntile * 16 + (lane & 15);
  const unsigned short* w1row = W1 + (size_t)n * 128;
  const unsigned short* w2row = W2 + (size_t)n * 128;

  v8f acc = {};
#pragma unroll
  for (int k0 = 0; k0 < 128; k0 += 32) {
    Frag16 a, b;
    a.q[0] = *(const uint4*)(a1row + k0 + half * 8);
    a.q[1] = *(const uint4*)(a1row + k0 + half * 8 + 16);
    b.q[0] = *(const uint4*)(w1row + k0 + half * 8);
    b.q[1] = *(const uint4*)(w1row + k0 + half * 8 + 16);
    acc = __builtin_amdgcn_wmma_f32_16x16x32_bf16(false, a.v, false, b.v,
                                                  (short)0, acc, false, false);
    a.q[0] = *(const uint4*)(a2row + k0 + half * 8);
    a.q[1] = *(const uint4*)(a2row + k0 + half * 8 + 16);
    b.q[0] = *(const uint4*)(w2row + k0 + half * 8);
    b.q[1] = *(const uint4*)(w2row + k0 + half * 8 + 16);
    acc = __builtin_amdgcn_wmma_f32_16x16x32_bf16(false, a.v, false, b.v,
                                                  (short)0, acc, false, false);
  }

  const float bv = bias[n];
#pragma unroll
  for (int r = 0; r < 8; ++r) {
    const int row = mtile * 16 + half * 8 + r;
    float v = acc[r] + bv;
    if (do_relu) v = v > 0.f ? v : 0.f;
    if (outb) outb[(size_t)row * NOUT + n] = (unsigned short)f32_to_bf16_rne(v);
    if (outf) outf[(size_t)row * NOUT + n] = v;
  }
}

// ---------------------------------------------------------------------------
// Degree count (once; dst-only dependent, shared by both layers)
// ---------------------------------------------------------------------------
__global__ void __launch_bounds__(256)
k_deg(const int* __restrict__ dst, float* __restrict__ deg, int E) {
  int t = blockIdx.x * 256 + threadIdx.x;
  if (t < E) gatomic_add_f32(&deg[dst[t]], 1.0f);
}

// ---------------------------------------------------------------------------
// Scatter-sum of bf16 messages into fp32 accumulator.
// One wave per edge, 4 features per lane (128 = 32 lanes * 4).
// Gather is one b64 per lane (full 256B row per wave, coalesced); the four
// adds go straight to the L2 atomic units.
// ---------------------------------------------------------------------------
__global__ void __launch_bounds__(256)
k_scatter(const unsigned short* __restrict__ xpb, const int* __restrict__ src,
          const int* __restrict__ dst, float* __restrict__ summed, int E) {
  int tid  = blockIdx.x * 256 + threadIdx.x;
  int edge = tid >> 5;
  int lane = tid & 31;
  if (edge >= E) return;
  int s = src[edge], d = dst[edge];
  const uint2 raw = *(const uint2*)(xpb + (size_t)s * 128 + lane * 4);
  float* o = summed + (size_t)d * 128 + lane * 4;
  gatomic_add_f32(o + 0, bf16_to_f32(raw.x & 0xFFFFu));
  gatomic_add_f32(o + 1, bf16_to_f32(raw.x >> 16));
  gatomic_add_f32(o + 2, bf16_to_f32(raw.y & 0xFFFFu));
  gatomic_add_f32(o + 3, bf16_to_f32(raw.y >> 16));
}

// ---------------------------------------------------------------------------
// agg_bf16 = bf16(summed / max(deg, 1)), 4 elems/thread.
// A quad never crosses a row (128 % 4 == 0), so one deg lookup per thread.
// ---------------------------------------------------------------------------
__global__ void __launch_bounds__(256)
k_div_deg(const float* __restrict__ summed, const float* __restrict__ deg,
          unsigned short* __restrict__ aggb, int nquads) {
  int i = blockIdx.x * 256 + threadIdx.x;
  if (i >= nquads) return;
  float d = deg[i >> 5];              // (i*4) >> 7
  d = d > 1.f ? d : 1.f;
  const float inv = 1.0f / d;
  const float4 v = *(const float4*)(summed + (size_t)i * 4);
  uint2 pk;
  pk.x = pack_bf16x2(v.x * inv, v.y * inv);
  pk.y = pack_bf16x2(v.z * inv, v.w * inv);
  *(uint2*)(aggb + (size_t)i * 4) = pk;
}

// ---------------------------------------------------------------------------
// Final epilogue per row of [N,64]: L2-normalize -> relu -> log_softmax.
// One wave per row, 2 elems per lane; wave32 shuffle reductions.
// ---------------------------------------------------------------------------
__global__ void __launch_bounds__(256)
k_final(float* __restrict__ out, int NN) {
  const int lane = threadIdx.x & 31;
  const int row  = blockIdx.x * 8 + (threadIdx.x >> 5);
  if (row >= NN) return;
  float* p = out + (size_t)row * 64;
  float x0 = p[lane * 2], x1 = p[lane * 2 + 1];

  float ss = x0 * x0 + x1 * x1;
#pragma unroll
  for (int o = 16; o > 0; o >>= 1) ss += __shfl_xor(ss, o, 32);
  float nrm = sqrtf(ss);
  nrm = nrm > 1e-12f ? nrm : 1e-12f;
  const float inv = 1.0f / nrm;
  x0 = fmaxf(x0 * inv, 0.f);
  x1 = fmaxf(x1 * inv, 0.f);

  float mx = fmaxf(x0, x1);
#pragma unroll
  for (int o = 16; o > 0; o >>= 1) mx = fmaxf(mx, __shfl_xor(mx, o, 32));
  float e = expf(x0 - mx) + expf(x1 - mx);
#pragma unroll
  for (int o = 16; o > 0; o >>= 1) e += __shfl_xor(e, o, 32);
  const float lse = mx + logf(e);

  p[lane * 2]     = x0 - lse;
  p[lane * 2 + 1] = x1 - lse;
}

// ---------------------------------------------------------------------------
// Host-side orchestration
// ---------------------------------------------------------------------------
extern "C" void kernel_launch(void* const* d_in, const int* in_sizes, int n_in,
                              void* d_out, int out_size, void* d_ws, size_t ws_size,
                              hipStream_t stream) {
  const int NN = in_sizes[0] / 128;   // 100000 nodes (multiple of 16)
  const int E  = in_sizes[1] / 2;     // 1600000 edges

  const float* x     = (const float*)d_in[0];
  const int*   ei    = (const int*)d_in[1];
  const float* p1_w  = (const float*)d_in[2];
  const float* p1_b  = (const float*)d_in[3];
  const float* l1_lw = (const float*)d_in[4];
  const float* l1_lb = (const float*)d_in[5];
  const float* l1_rw = (const float*)d_in[6];
  const float* p2_w  = (const float*)d_in[7];
  const float* p2_b  = (const float*)d_in[8];
  const float* l2_lw = (const float*)d_in[9];
  const float* l2_lb = (const float*)d_in[10];
  const float* l2_rw = (const float*)d_in[11];

  const int* src = ei;
  const int* dst = ei + E;

  // Workspace bump allocator (256B aligned regions)
  char* wsp = (char*)d_ws;
  auto alloc = [&](size_t bytes) -> char* {
    char* p = wsp;
    wsp += (bytes + 255) & ~(size_t)255;
    return p;
  };
  unsigned short* xbf   = (unsigned short*)alloc((size_t)NN * 128 * 2); // reused as h_bf16
  unsigned short* p1wb  = (unsigned short*)alloc(128 * 128 * 2);
  unsigned short* l1lwb = (unsigned short*)alloc(128 * 128 * 2);
  unsigned short* l1rwb = (unsigned short*)alloc(128 * 128 * 2);
  unsigned short* p2wb  = (unsigned short*)alloc(128 * 128 * 2);
  unsigned short* l2lwb = (unsigned short*)alloc(64 * 128 * 2);
  unsigned short* l2rwb = (unsigned short*)alloc(64 * 128 * 2);
  unsigned short* xpb   = (unsigned short*)alloc((size_t)NN * 128 * 2);
  float*          summ  = (float*)alloc((size_t)NN * 128 * 4);
  float*          deg   = (float*)alloc((size_t)NN * 4);
  unsigned short* aggb  = (unsigned short*)alloc((size_t)NN * 128 * 2);
  unsigned short* hbf   = xbf;   // x is dead after layer-1 projection

  auto cvt = [&](const float* in, unsigned short* out, int n) {
    const int q = n / 4;
    k_f32_to_bf16<<<(q + 255) / 256, 256, 0, stream>>>(in, out, q);
  };

  // --- precision conversion (bf16 GEMM path) ---
  cvt(x,     xbf,   NN * 128);
  cvt(p1_w,  p1wb,  128 * 128);
  cvt(l1_lw, l1lwb, 128 * 128);
  cvt(l1_rw, l1rwb, 128 * 128);
  cvt(p2_w,  p2wb,  128 * 128);
  cvt(l2_lw, l2lwb, 64 * 128);
  cvt(l2_rw, l2rwb, 64 * 128);

  // --- degrees (shared by both layers) ---
  hipMemsetAsync(deg, 0, (size_t)NN * 4, stream);
  k_deg<<<(E + 255) / 256, 256, 0, stream>>>(dst, deg, E);

  const int mtiles = NN / 16;
  const int scatter_blocks = (E * 32 + 255) / 256;
  const int quads = NN * 32;          // (NN*128)/4

  // --- layer 1 ---
  k_gemm_proj<<<mtiles, 256, 0, stream>>>(xbf, p1wb, p1_b, xpb);
  hipMemsetAsync(summ, 0, (size_t)NN * 128 * 4, stream);
  k_scatter<<<scatter_blocks, 256, 0, stream>>>(xpb, src, dst, summ, E);
  k_div_deg<<<(quads + 255) / 256, 256, 0, stream>>>(summ, deg, aggb, quads);
  k_dual_gemm<8><<<mtiles, 256, 0, stream>>>(aggb, l1lwb, xpb, l1rwb, l1_lb,
                                             hbf, nullptr, /*relu=*/1);

  // --- layer 2 (buffers reused) ---
  k_gemm_proj<<<mtiles, 256, 0, stream>>>(hbf, p2wb, p2_b, xpb);
  hipMemsetAsync(summ, 0, (size_t)NN * 128 * 4, stream);
  k_scatter<<<scatter_blocks, 256, 0, stream>>>(xpb, src, dst, summ, E);
  k_div_deg<<<(quads + 255) / 256, 256, 0, stream>>>(summ, deg, aggb, quads);
  k_dual_gemm<4><<<mtiles, 128, 0, stream>>>(aggb, l2lwb, xpb, l2rwb, l2_lb,
                                             nullptr, (float*)d_out, /*relu=*/0);

  // --- normalize -> relu -> log_softmax, in place on d_out ---
  k_final<<<(NN + 7) / 8, 256, 0, stream>>>((float*)d_out, NN);
}
